// LabelEnhanceBert_2001454760481
// MI455X (gfx1250) — compile-verified, matrix operand order checked by reading, not run on hardware
//
#include <hip/hip_runtime.h>
#include <hip/hip_bf16.h>

typedef __bf16 bf16_t;
typedef __attribute__((ext_vector_type(16))) __bf16 v16bf;
typedef __attribute__((ext_vector_type(8)))  __bf16 v8bf;
typedef __attribute__((ext_vector_type(8)))  float  v8f;

#define H_DIM   1792
#define B_DIM   4
#define L1_DIM  128
#define NL_DIM  36
#define L2_DIM  16
#define DB_DIM  768
#define DE_DIM  512
#define M_T     (B_DIM * L1_DIM)    // 512 text rows
#define M_L     (NL_DIM * L2_DIM)   // 576 label rows

// ---------------------------------------------------------------------------
// Fragment loader for 16-bit WMMA operands (A: MxK row-major, B: W[N][K]
// row-major == KxN column-access). Per CDNA5 ISA 16-bit layout:
// lanes 0-15 hold K = {k0..k0+7, k0+16..k0+23}, lanes 16-31 hold
// K = {k0+8..k0+15, k0+24..k0+31}, each lane owning one row/column.
// Two 16-byte loads per fragment.
// ---------------------------------------------------------------------------
__device__ inline v16bf load_frag(const bf16_t* base, int stride, int row,
                                  int k0, int lane) {
  const int half = (lane >> 4) & 1;
  const bf16_t* p = base + (size_t)row * stride + k0 + half * 8;
  v8bf lo = *(const v8bf*)(p);
  v8bf hi = *(const v8bf*)(p + 16);
  v16bf r;
#pragma unroll
  for (int i = 0; i < 8; ++i) { r[i] = lo[i]; r[i + 8] = hi[i]; }
  return r;
}

__device__ inline v8f wmma_bf16(v16bf a, v16bf b, v8f c) {
  return __builtin_amdgcn_wmma_f32_16x16x32_bf16(
      /*neg_a=*/false, a, /*neg_b=*/false, b,
      /*c_mod=*/(short)0, c, /*reuse_a=*/false, /*reuse_b=*/false);
}

// ---------------------------------------------------------------------------
// Build enc (concat of bert + pos_emb gather + ent_emb gather) as bf16.
// ---------------------------------------------------------------------------
__global__ void build_enc_kernel(const float* __restrict__ bert,
                                 const int* __restrict__ posTok,
                                 const int* __restrict__ entTok,
                                 const float* __restrict__ pos_emb,
                                 const float* __restrict__ ent_emb,
                                 bf16_t* __restrict__ out, int rows) {
  int i = blockIdx.x * blockDim.x + threadIdx.x;
  int total = rows * H_DIM;
  if (i >= total) return;
  int r = i / H_DIM;
  int h = i - r * H_DIM;
  float v;
  if (h < DB_DIM) {
    v = bert[(size_t)r * DB_DIM + h];
  } else if (h < DB_DIM + DE_DIM) {
    v = pos_emb[(size_t)posTok[r] * DE_DIM + (h - DB_DIM)];
  } else {
    v = ent_emb[(size_t)entTok[r] * DE_DIM + (h - DB_DIM - DE_DIM)];
  }
  out[i] = (bf16_t)v;
}

// f32 -> bf16 conversion (weights).
__global__ void cvt_bf16_kernel(const float* __restrict__ in,
                                bf16_t* __restrict__ out, int n) {
  int i = blockIdx.x * blockDim.x + threadIdx.x;
  if (i < n) out[i] = (bf16_t)in[i];
}

// ---------------------------------------------------------------------------
// Generic WMMA GEMM: C[M][N] (f32, optional bf16 copy) = A[M][K] * W[N][K]^T.
// Block = 256 threads = 8 waves as 4(m-strips of 16) x 2(n-halves of 32).
// Tile = 64x64, K streamed in steps of 32.
// ---------------------------------------------------------------------------
__global__ void gemm_bf16_kernel(const bf16_t* __restrict__ A,
                                 const bf16_t* __restrict__ W,
                                 float* __restrict__ C,
                                 bf16_t* __restrict__ Cb,
                                 int M, int N, int K) {
  const int lane = threadIdx.x & 31;
  const int w    = threadIdx.x >> 5;
  const int wm   = w & 3;
  const int wn   = w >> 2;
  const int m0   = blockIdx.y * 64 + wm * 16;
  const int n0   = blockIdx.x * 64 + wn * 32;

  const int arow  = m0 + (lane & 15);
  const int bcol0 = n0 + (lane & 15);
  const int bcol1 = bcol0 + 16;

  v8f acc0 = {}; v8f acc1 = {};
  for (int k0 = 0; k0 < K; k0 += 32) {
    __builtin_prefetch((const void*)(W + (size_t)bcol0 * K + k0 + 512), 0, 0);
    v16bf a  = load_frag(A, K, arow,  k0, lane);
    v16bf b0 = load_frag(W, K, bcol0, k0, lane);
    v16bf b1 = load_frag(W, K, bcol1, k0, lane);
    acc0 = wmma_bf16(a, b0, acc0);
    acc1 = wmma_bf16(a, b1, acc1);
  }

  // D layout: VGPR j -> (M = m0 + j + 8*(lane>=16), N = n-base + (lane&15)).
  const int mrow = m0 + ((lane >> 4) & 1) * 8;
  const int ncol = n0 + (lane & 15);
#pragma unroll
  for (int j = 0; j < 8; ++j) {
    size_t i0 = (size_t)(mrow + j) * N + ncol;
    C[i0]      = acc0[j];
    C[i0 + 16] = acc1[j];
    if (Cb) {
      Cb[i0]      = (bf16_t)acc0[j];
      Cb[i0 + 16] = (bf16_t)acc1[j];
    }
  }
}

// ---------------------------------------------------------------------------
// Softmax over L2=16 per (bl, n) group, with attention-mask additive term.
// ---------------------------------------------------------------------------
__global__ void softmax16_kernel(float* __restrict__ SC,
                                 const int* __restrict__ mask) {
  int g = blockIdx.x * blockDim.x + threadIdx.x;
  if (g >= M_T * NL_DIM) return;
  int bl = g / NL_DIM;
  int n  = g - bl * NL_DIM;
  float* p = SC + (size_t)bl * M_L + n * L2_DIM;
  const int* mr = mask + n * L2_DIM;
  float v[L2_DIM];
  float mx = -1e30f;
#pragma unroll
  for (int k = 0; k < L2_DIM; ++k) {
    float s = p[k] + (1.0f - (float)mr[k]) * -10000.0f;
    v[k] = s;
    mx = fmaxf(mx, s);
  }
  float sum = 0.0f;
#pragma unroll
  for (int k = 0; k < L2_DIM; ++k) { v[k] = __expf(v[k] - mx); sum += v[k]; }
  float inv = 1.0f / sum;
#pragma unroll
  for (int k = 0; k < L2_DIM; ++k) p[k] = v[k] * inv;
}

// ---------------------------------------------------------------------------
// Fused final stage (dominant: 118 GFLOP). Block = (n, 64-row bl tile).
//
// Phase 1: fused[m][h] = TF[bl][h] + sum_k SC[bl][n*16+k] * LF[n*16+k][h],
//          stored bf16 in a 64x1792 DYNAMIC-LDS tile (226 KB; CDNA5 allows
//          up to 320 KB per workgroup).
// Phase 2: WMMA against W3 over all 1792 output columns. 8 waves =
//          4 m-strips x 2 o-groups; each wave amortizes ONE LDS A-fragment
//          over FOUR B fragments / WMMAs per K-step, then applies
//          tanh(.+b3) and reduces directly into start/end logits.
// W3 L2 traffic: 6.4 MB x 288 blocks = 1.85 GB at 64 FLOP/byte.
// ---------------------------------------------------------------------------
#define FSTRIDE 1808  // 1792 + 16 pad, keeps 16B alignment

__global__ void fused_w3_logits_kernel(const float* __restrict__ TF,
                                       const float* __restrict__ LF,
                                       const float* __restrict__ SC,
                                       const bf16_t* __restrict__ W3b,
                                       const float* __restrict__ b3,
                                       const float* __restrict__ w_start,
                                       const float* __restrict__ b_start,
                                       const float* __restrict__ w_end,
                                       const float* __restrict__ b_end,
                                       float* __restrict__ outStart,
                                       float* __restrict__ outEnd) {
  extern __shared__ bf16_t sF[];        // 64 * FSTRIDE bf16 (dynamic LDS)
  __shared__ float sS[64 * L2_DIM];     // scores tile
  __shared__ float sRed[64];
  __shared__ float eRed[64];

  const int n   = blockIdx.x;       // label index 0..35
  const int bl0 = blockIdx.y * 64;  // 64 (b,l) rows per block
  const int tid = threadIdx.x;

  // Stage the 64x16 softmaxed score tile; zero reduction slots.
  for (int i = tid; i < 64 * L2_DIM; i += 256) {
    int m = i >> 4;
    int k = i & 15;
    sS[i] = SC[(size_t)(bl0 + m) * M_L + n * L2_DIM + k];
  }
  if (tid < 64) { sRed[tid] = 0.0f; eRed[tid] = 0.0f; }
  __syncthreads();

  // Phase 1: build fused tile (VALU; K=16 weighted sum is cheap).
  for (int e = tid; e < 64 * H_DIM; e += 256) {
    int m = e / H_DIM;
    int h = e - m * H_DIM;
    float acc = TF[(size_t)(bl0 + m) * H_DIM + h];
    const float* srow = &sS[m * L2_DIM];
    const float* lcol = LF + (size_t)(n * L2_DIM) * H_DIM + h;
#pragma unroll
    for (int k = 0; k < L2_DIM; ++k) acc += srow[k] * lcol[(size_t)k * H_DIM];
    sF[m * FSTRIDE + h] = (bf16_t)acc;
  }
  __syncthreads();

  // Phase 2.
  const int lane = tid & 31;
  const int w    = tid >> 5;
  const int ms   = w & 3;   // m-strip 0..3
  const int og   = w >> 2;  // o-group 0..1 (64 cols each within 128 window)
  const int arow = ms * 16 + (lane & 15);

  float sAcc[8] = {0.f, 0.f, 0.f, 0.f, 0.f, 0.f, 0.f, 0.f};
  float eAcc[8] = {0.f, 0.f, 0.f, 0.f, 0.f, 0.f, 0.f, 0.f};

  for (int oblk = 0; oblk < H_DIM; oblk += 128) {
    const int obase = oblk + og * 64;
    v8f acc[4];
    acc[0] = v8f{}; acc[1] = v8f{}; acc[2] = v8f{}; acc[3] = v8f{};
    for (int k0 = 0; k0 < H_DIM; k0 += 32) {
      __builtin_prefetch(
          (const void*)(W3b + (size_t)(obase + (lane & 15)) * H_DIM + k0 + 512),
          0, 0);
      v16bf a = load_frag(sF, FSTRIDE, arow, k0, lane);  // 1 LDS frag ...
#pragma unroll
      for (int s = 0; s < 4; ++s) {                       // ... 4 WMMAs
        int ocol = obase + s * 16 + (lane & 15);
        v16bf b = load_frag(W3b, H_DIM, ocol, k0, lane);
        acc[s] = wmma_bf16(a, b, acc[s]);
      }
    }
#pragma unroll
    for (int s = 0; s < 4; ++s) {
      const int ocol = obase + s * 16 + (lane & 15);
      const float bb  = b3[ocol];
      const float wsv = w_start[ocol];
      const float wev = w_end[ocol];
#pragma unroll
      for (int j = 0; j < 8; ++j) {
        float y = tanhf(acc[s][j] + bb);
        sAcc[j] += y * wsv;
        eAcc[j] += y * wev;
      }
    }
  }

  // Reduce over the 16 lanes of each half (each lane held distinct ocols),
  // then across waves via LDS atomics.
#pragma unroll
  for (int j = 0; j < 8; ++j) {
    float s = sAcc[j];
    float e = eAcc[j];
#pragma unroll
    for (int m2 = 1; m2 < 16; m2 <<= 1) {
      s += __shfl_xor(s, m2, 32);
      e += __shfl_xor(e, m2, 32);
    }
    if ((lane & 15) == 0) {
      int m = ms * 16 + j + ((lane >> 4) & 1) * 8;  // D layout row
      atomicAdd(&sRed[m], s);
      atomicAdd(&eRed[m], e);
    }
  }
  __syncthreads();

  if (tid < 64) {
    int bl = bl0 + tid;
    outStart[(size_t)bl * NL_DIM + n] = sRed[tid] + b_start[0];
    outEnd  [(size_t)bl * NL_DIM + n] = eRed[tid] + b_end[0];
  }
}

// ---------------------------------------------------------------------------
// Launcher
// ---------------------------------------------------------------------------
extern "C" void kernel_launch(void* const* d_in, const int* in_sizes, int n_in,
                              void* d_out, int out_size, void* d_ws, size_t ws_size,
                              hipStream_t stream) {
  const float* text_bert  = (const float*)d_in[0];
  const float* label_bert = (const float*)d_in[1];
  const int*   text_pos   = (const int*)d_in[2];
  const int*   text_ent   = (const int*)d_in[3];
  const int*   label_pos  = (const int*)d_in[4];
  const int*   label_ent  = (const int*)d_in[5];
  const int*   label_mask = (const int*)d_in[6];
  const float* pos_emb    = (const float*)d_in[7];
  const float* ent_emb    = (const float*)d_in[8];
  const float* W1         = (const float*)d_in[9];
  const float* W2         = (const float*)d_in[10];
  const float* W3         = (const float*)d_in[11];
  const float* b3         = (const float*)d_in[12];
  const float* w_start    = (const float*)d_in[13];
  const float* b_start    = (const float*)d_in[14];
  const float* w_end      = (const float*)d_in[15];
  const float* b_end      = (const float*)d_in[16];

  // Workspace carve-out (256B aligned).
  char* ws = (char*)d_ws;
  size_t off = 0;
  auto carve = [&](size_t bytes) -> char* {
    char* p = ws + off;
    off = (off + bytes + 255) & ~(size_t)255;
    return p;
  };
  bf16_t* encTb = (bf16_t*)carve((size_t)M_T * H_DIM * 2);
  bf16_t* encLb = (bf16_t*)carve((size_t)M_L * H_DIM * 2);
  bf16_t* W1b   = (bf16_t*)carve((size_t)H_DIM * H_DIM * 2);
  bf16_t* W2b   = (bf16_t*)carve((size_t)H_DIM * H_DIM * 2);
  bf16_t* W3b   = (bf16_t*)carve((size_t)H_DIM * H_DIM * 2);
  float*  TF    = (float*) carve((size_t)M_T * H_DIM * 4);
  bf16_t* TFb   = (bf16_t*)carve((size_t)M_T * H_DIM * 2);
  float*  LF    = (float*) carve((size_t)M_L * H_DIM * 4);
  bf16_t* LFb   = (bf16_t*)carve((size_t)M_L * H_DIM * 2);
  float*  SC    = (float*) carve((size_t)M_T * M_L * 4);
  (void)ws_size;

  // 1) Build bf16 enc tensors (concat + embedding gathers).
  {
    int tot = M_T * H_DIM;
    build_enc_kernel<<<(tot + 255) / 256, 256, 0, stream>>>(
        text_bert, text_pos, text_ent, pos_emb, ent_emb, encTb, M_T);
    tot = M_L * H_DIM;
    build_enc_kernel<<<(tot + 255) / 256, 256, 0, stream>>>(
        label_bert, label_pos, label_ent, pos_emb, ent_emb, encLb, M_L);
  }
  // 2) Convert weights to bf16.
  {
    int n = H_DIM * H_DIM;
    int g = (n + 255) / 256;
    cvt_bf16_kernel<<<g, 256, 0, stream>>>(W1, W1b, n);
    cvt_bf16_kernel<<<g, 256, 0, stream>>>(W2, W2b, n);
    cvt_bf16_kernel<<<g, 256, 0, stream>>>(W3, W3b, n);
  }
  // 3) token_feature = encT @ W1^T, label_feature = encL @ W2^T.
  gemm_bf16_kernel<<<dim3(H_DIM / 64, M_T / 64), 256, 0, stream>>>(
      encTb, W1b, TF, TFb, M_T, H_DIM, H_DIM);
  gemm_bf16_kernel<<<dim3(H_DIM / 64, M_L / 64), 256, 0, stream>>>(
      encLb, W2b, LF, LFb, M_L, H_DIM, H_DIM);
  // 4) scores = TF @ LF^T.
  gemm_bf16_kernel<<<dim3(M_L / 64, M_T / 64), 256, 0, stream>>>(
      TFb, LFb, SC, (bf16_t*)nullptr, M_T, M_L, H_DIM);
  // 5) masked softmax over L2 groups.
  softmax16_kernel<<<(M_T * NL_DIM + 255) / 256, 256, 0, stream>>>(SC, label_mask);
  // 6) fused tile + W3 GEMM + tanh + logit reductions (dominant stage).
  //    64x1792 bf16 fused tile in dynamic LDS (226 KB of the WGP's 320 KB).
  float* outStart = (float*)d_out;
  float* outEnd   = outStart + (size_t)M_T * NL_DIM;
  size_t fused_lds = (size_t)64 * FSTRIDE * sizeof(bf16_t);
  fused_w3_logits_kernel<<<dim3(NL_DIM, M_T / 64), 256, fused_lds, stream>>>(
      TF, LF, SC, W3b, b3, w_start, b_start, w_end, b_end, outStart, outEnd);
}
